// RNNRegression_26216480375393
// MI455X (gfx1250) — matrix operand order repaired and checked by reading, not compiled, not run on hardware
//
#include <hip/hip_runtime.h>

// ---------------------------------------------------------------------------
// RNN regression for MI455X (gfx1250, wave32, WMMA bf16, f32 accumulate).
// B=256, T=512, I=256, H=256.
//
// Fast path (needs 134 MB scratch for xw = x@W_ih^T + bias):
//   K1: whole-chip WMMA GEMM, 131072x256x256 -> xw (f32, stays L2-resident).
//   K2: serial recurrence, 16 blocks x 16 waves; per step per wave only
//       8 WMMAs (h @ W_hh^T), W_hh fragments register-resident for all 512
//       steps, A-fragments double-buffered from LDS, 1 barrier per step,
//       xw(t+1) prefetched to registers during step t.
// Fallback (no scratch): fused kernel, x-projection WMMAs in the loop with
//   W_ih staged in LDS, W_hh still register-resident.
// ---------------------------------------------------------------------------

typedef __bf16 bf16_t;
typedef __attribute__((ext_vector_type(16))) __bf16 v16bf;
typedef __attribute__((ext_vector_type(8)))  __bf16 v8bf;
typedef __attribute__((ext_vector_type(8)))  float  v8f;

#define B_  256
#define T_  512
#define I_  256
#define H_  256
#define BM  16
#define NWAVES 16
#define NTHREADS (NWAVES * 32)

union V16U { v16bf v; v8bf h[2]; };
union V8FU { v8f v; float e[8]; };
struct alignas(8) bf4 { bf16_t e[4]; };

#define WMMA_BF16(a, b, c) __builtin_amdgcn_wmma_f32_16x16x32_bf16( \
    false, (a), false, (b), (short)0, (c), false, false)

__device__ inline v8bf cvt8(float4 q0, float4 q1) {
    union { v8bf v; bf16_t e[8]; } s;
    s.e[0] = (bf16_t)q0.x; s.e[1] = (bf16_t)q0.y;
    s.e[2] = (bf16_t)q0.z; s.e[3] = (bf16_t)q0.w;
    s.e[4] = (bf16_t)q1.x; s.e[5] = (bf16_t)q1.y;
    s.e[6] = (bf16_t)q1.z; s.e[7] = (bf16_t)q1.w;
    return s.v;
}

// B-fragment (two 8-elem K runs) from a global f32 weight row, cvt->bf16.
__device__ inline v16bf load_wfrag(const float* __restrict__ wrow, int kb, int hi) {
    const float* p = wrow + kb * 32 + hi * 8;
    V16U w;
    w.h[0] = cvt8(*(const float4*)(p),     *(const float4*)(p + 4));
    w.h[1] = cvt8(*(const float4*)(p + 16), *(const float4*)(p + 20));
    return w.v;
}
// A-fragment (two 8-elem K runs) from a global f32 row, cvt->bf16.
__device__ inline v16bf load_afrag_g(const float* __restrict__ arow, int kb, int hi) {
    const float* p = arow + kb * 32 + hi * 8;
    V16U a;
    a.h[0] = cvt8(*(const float4*)(p),      *(const float4*)(p + 4));
    a.h[1] = cvt8(*(const float4*)(p + 16), *(const float4*)(p + 20));
    return a.v;
}
// A/B fragments from LDS bf16 rows.
__device__ inline v16bf ldsA(const bf16_t* row, int kb, int hi) {
    V16U a;
    a.h[0] = *(const v8bf*)(row + kb * 32 + hi * 8);
    a.h[1] = *(const v8bf*)(row + kb * 32 + hi * 8 + 16);
    return a.v;
}
__device__ inline v16bf ldsB(const bf16_t* row, int kb, int hi) {
    V16U b;
    b.h[0] = *(const v8bf*)(row + kb * 32 + hi * 8);
    b.h[1] = *(const v8bf*)(row + kb * 32 + hi * 8 + 16);
    return b.v;
}

// ---------------------------------------------------------------------------
// K1: xw[b*T+t, n] = sum_k x[b,t,k] * W_ih[n,k] + b_ih[n] + b_hh[n]
// One 16x16 tile per wave, 8 waves/block, 16384 blocks. All-global, no LDS.
// ---------------------------------------------------------------------------
extern "C" __global__ __launch_bounds__(256)
void xw_precompute(const float* __restrict__ x,
                   const float* __restrict__ W_ih,
                   const float* __restrict__ b_ih,
                   const float* __restrict__ b_hh,
                   float* __restrict__ xw)
{
    const int lane = threadIdx.x & 31;
    const int wv   = threadIdx.x >> 5;
    const int tile = blockIdx.x * 8 + wv;       // 0 .. 131071
    const int mt   = tile >> 4;                 // M tile (B*T)/16
    const int nt   = tile & 15;                 // N tile
    const int hi   = lane >> 4;
    const int nn   = lane & 15;

    const float* arow = x    + (size_t)(mt * 16 + nn) * I_;
    const float* brow = W_ih + (size_t)(nt * 16 + nn) * I_;
    const float  bsc  = b_ih[nt * 16 + nn] + b_hh[nt * 16 + nn];

    V8FU acc;
    #pragma unroll
    for (int r = 0; r < 8; ++r) acc.e[r] = bsc;
    #pragma unroll
    for (int kb = 0; kb < I_ / 32; ++kb) {
        v16bf a = load_afrag_g(arow, kb, hi);
        v16bf b = load_wfrag(brow, kb, hi);
        acc.v = WMMA_BF16(a, b, acc.v);
    }
    float* orow = xw + (size_t)(mt * 16 + hi * 8) * H_ + nt * 16 + nn;
    #pragma unroll
    for (int r = 0; r < 8; ++r) orow[(size_t)r * H_] = acc.e[r];
}

// ---------------------------------------------------------------------------
// K2: recurrence. PRE=true reads precomputed xw; PRE=false fuses x-projection.
// ---------------------------------------------------------------------------
template<bool PRE>
__global__ __launch_bounds__(NTHREADS)
void rnn_rec(const float* __restrict__ x,
             const float* __restrict__ xw,
             const float* __restrict__ W_ih,
             const float* __restrict__ W_hh,
             const float* __restrict__ b_ih,
             const float* __restrict__ b_hh,
             const float* __restrict__ W_fc,
             const float* __restrict__ b_fc,
             float* __restrict__ out)
{
    extern __shared__ char smem[];
    bf16_t* wih  = (bf16_t*)smem;                         // fused only [H_][I_]
    char*   rest = smem + (PRE ? 0 : (size_t)H_ * I_ * 2);
    bf16_t* hls0 = (bf16_t*)rest;                         // [BM][H_] ping
    bf16_t* hls1 = hls0 + BM * H_;                        // pong
    float*  hfin = (float*)(hls1 + BM * H_);              // [BM][H_] f32
    bf16_t* xls0 = (bf16_t*)(hfin + BM * H_);             // fused only
    bf16_t* xls1 = xls0 + BM * I_;

    const int tid  = threadIdx.x;
    const int lane = tid & 31;
    const int wv   = tid >> 5;
    const int hi   = lane >> 4;
    const int nn   = lane & 15;
    const int n    = wv * 16 + nn;          // this lane's hidden column
    const int mrow = hi * 8;                // C/D row base
    const int b0   = blockIdx.x * BM;

    // ---- init ----
    if (!PRE) {
        for (int i = tid; i < (H_ * I_) / 4; i += NTHREADS) {
            float4 a = ((const float4*)W_ih)[i];
            bf4 pa = { (bf16_t)a.x, (bf16_t)a.y, (bf16_t)a.z, (bf16_t)a.w };
            ((bf4*)wih)[i] = pa;
        }
    }
    for (int i = tid; i < 2 * BM * H_; i += NTHREADS)
        hls0[i] = (bf16_t)0.0f;             // zeros hls0 + hls1 (contiguous)

    // W_hh fragments register-resident for the entire 512-step loop (64 VGPRs)
    V16U whhf[8];
    {
        const float* whrow = W_hh + (size_t)n * H_;
        #pragma unroll
        for (int kb = 0; kb < 8; ++kb) whhf[kb].v = load_wfrag(whrow, kb, hi);
    }
    const float bsc = PRE ? 0.0f : (b_ih[n] + b_hh[n]);

    // fused-path x staging: thread covers 8 contiguous floats of [BM][I_]
    const int xrow = (tid * 8) >> 8;
    const int xcol = (tid * 8) & (I_ - 1);
    const float* xgp = x + (size_t)(b0 + xrow) * T_ * I_ + xcol;
    float4 p0, p1;
    // PRE-path xw prefetch registers
    const float* xb = xw + (size_t)(b0 + mrow) * T_ * H_ + n;
    float pw[8];

    if (PRE) {
        #pragma unroll
        for (int r = 0; r < 8; ++r) pw[r] = xb[(size_t)r * T_ * H_];   // t = 0
    } else {
        p0 = *(const float4*)(xgp);
        p1 = *(const float4*)(xgp + 4);
    }
    __syncthreads();

    const bf16_t* brow_ih = wih + (size_t)n * I_;
    V8FU acc;

    for (int t = 0; t < T_; ++t) {
        const bf16_t* hr = (t & 1) ? hls1 : hls0;
        bf16_t*       hw = (t & 1) ? hls0 : hls1;

        if (PRE) {
            #pragma unroll
            for (int r = 0; r < 8; ++r) acc.e[r] = pw[r];
            if (t + 1 < T_) {               // prefetch xw(t+1), L2-resident
                #pragma unroll
                for (int r = 0; r < 8; ++r)
                    pw[r] = xb[((size_t)r * T_ + (t + 1)) * H_];
            }
        } else {
            bf16_t* xw_buf = (t & 1) ? xls1 : xls0;
            union { v8bf v; bf16_t e[8]; } s;
            s.e[0] = (bf16_t)p0.x; s.e[1] = (bf16_t)p0.y;
            s.e[2] = (bf16_t)p0.z; s.e[3] = (bf16_t)p0.w;
            s.e[4] = (bf16_t)p1.x; s.e[5] = (bf16_t)p1.y;
            s.e[6] = (bf16_t)p1.z; s.e[7] = (bf16_t)p1.w;
            *(v8bf*)(xw_buf + xrow * I_ + xcol) = s.v;
            if (t + 1 < T_) {
                const float* xq = xgp + (size_t)(t + 1) * I_;
                p0 = *(const float4*)(xq);
                p1 = *(const float4*)(xq + 4);
            }
            #pragma unroll
            for (int r = 0; r < 8; ++r) acc.e[r] = bsc;
        }

        __syncthreads();   // h(t-1) (and staged x_t) visible; 1 barrier/step

        const bf16_t* harow = hr + nn * H_;
        V16U ab[2], bb[2];

        if (!PRE) {
            // input projection: A from staged x_t, B from W_ih in LDS
            const bf16_t* xarow = ((t & 1) ? xls1 : xls0) + nn * I_;
            ab[0].v = ldsA(xarow, 0, hi);
            bb[0].v = ldsB(brow_ih, 0, hi);
            #pragma unroll
            for (int kb = 0; kb < 8; ++kb) {
                if (kb < 7) {
                    ab[(kb + 1) & 1].v = ldsA(xarow, kb + 1, hi);
                    bb[(kb + 1) & 1].v = ldsB(brow_ih, kb + 1, hi);
                } else {
                    ab[(kb + 1) & 1].v = ldsA(harow, 0, hi);  // head of h GEMM
                }
                acc.v = WMMA_BF16(ab[kb & 1].v, bb[kb & 1].v, acc.v);
            }
        } else {
            ab[0].v = ldsA(harow, 0, hi);
        }

        // recurrent GEMM: A double-buffered from LDS, B register-resident
        #pragma unroll
        for (int kb = 0; kb < 8; ++kb) {
            if (kb < 7) ab[(kb + 1) & 1].v = ldsA(harow, kb + 1, hi);
            acc.v = WMMA_BF16(ab[kb & 1].v, whhf[kb].v, acc.v);
        }

        // tanh (overflow-safe, v_exp_f32 based)
        #pragma unroll
        for (int r = 0; r < 8; ++r) {
            float v = acc.e[r];
            float e = __expf(2.0f * fabsf(v));
            float tt = 1.0f - __fdividef(2.0f, e + 1.0f);
            acc.e[r] = copysignf(tt, v);
        }
        #pragma unroll
        for (int r = 0; r < 8; ++r)
            hw[(mrow + r) * H_ + n] = (bf16_t)acc.e[r];
    }

    // ---- head: out[b] = h_T[b,:] . W_fc + b_fc via f32 LDS stage ----
    __syncthreads();
    {
        const float wf = W_fc[n];
        #pragma unroll
        for (int r = 0; r < 8; ++r)
            hfin[(mrow + r) * H_ + n] = acc.e[r] * wf;
    }
    __syncthreads();
    if (tid < BM) {
        const float4* rowp = (const float4*)(hfin + tid * H_);
        float s = 0.0f;
        for (int i = 0; i < H_ / 4; ++i) {
            float4 q = rowp[i];
            s += q.x + q.y + q.z + q.w;
        }
        out[b0 + tid] = s + b_fc[0];
    }
}

#define SMEM_PRE   ((2 * BM * H_) * 2 + BM * H_ * 4)                       // 32 KB
#define SMEM_FUSED (H_ * I_ * 2 + SMEM_PRE + (2 * BM * I_) * 2)            // 176 KB
static_assert(SMEM_FUSED <= 320 * 1024, "LDS budget exceeded");

extern "C" void kernel_launch(void* const* d_in, const int* in_sizes, int n_in,
                              void* d_out, int out_size, void* d_ws, size_t ws_size,
                              hipStream_t stream) {
    const float* x    = (const float*)d_in[0];
    const float* W_ih = (const float*)d_in[1];
    const float* W_hh = (const float*)d_in[2];
    const float* b_ih = (const float*)d_in[3];
    const float* b_hh = (const float*)d_in[4];
    const float* W_fc = (const float*)d_in[5];
    const float* b_fc = (const float*)d_in[6];
    float* out = (float*)d_out;
    (void)in_sizes; (void)n_in; (void)out_size;

    const size_t XW_BYTES = (size_t)B_ * T_ * H_ * sizeof(float);  // 134 MB

    if (ws_size >= XW_BYTES) {
        float* xw = (float*)d_ws;
        xw_precompute<<<(B_ * T_ / 16) * (H_ / 16) / 8, 256, 0, stream>>>(
            x, W_ih, b_ih, b_hh, xw);
        hipFuncSetAttribute(reinterpret_cast<const void*>(&rnn_rec<true>),
                            hipFuncAttributeMaxDynamicSharedMemorySize, SMEM_PRE);
        rnn_rec<true><<<B_ / BM, NTHREADS, SMEM_PRE, stream>>>(
            x, xw, W_ih, W_hh, b_ih, b_hh, W_fc, b_fc, out);
    } else {
        hipFuncSetAttribute(reinterpret_cast<const void*>(&rnn_rec<false>),
                            hipFuncAttributeMaxDynamicSharedMemorySize, SMEM_FUSED);
        rnn_rec<false><<<B_ / BM, NTHREADS, SMEM_FUSED, stream>>>(
            x, nullptr, W_ih, W_hh, b_ih, b_hh, W_fc, b_fc, out);
    }
}
